// Mamba_8589935222
// MI455X (gfx1250) — compile-verified, matrix operand order checked by reading, not run on hardware
//
#include <hip/hip_runtime.h>

#define D_MODEL 1024
#define N_LAYERS 4
#define VOCAB   32000
#define D_STATE 16
#define D_INNER 2048
#define D_CONV  4
#define BATCH   2
#define SEQ     1024
#define ROWS    (BATCH * SEQ)   // 2048 token rows

typedef __attribute__((ext_vector_type(16))) _Float16 v16h;
typedef __attribute__((ext_vector_type(8)))  float    v8f;

// ---------------------------------------------------------------------------
// WMMA GEMM:  C[M,N] = A[M,K] @ W[N,K]^T (+bias[n]) (silu?) (+=C?)
// Block = 256 threads (8 wave32). Block tile 128(M) x 128(N), K-chunks of 32.
// Wave w owns the 16-row strip [w*16, w*16+16) x 128 cols:
//   1 shared A fragment + 8 B fragments -> 8 WMMAs per K-step.
// ---------------------------------------------------------------------------
#define BLK_M 128
#define BLK_N 128
#define BK    32
#define LDT   40   // LDS row stride in halfs (multiple of 4 -> 8B-aligned rows)

union Frag16 { v16h v; unsigned u[8]; };
union Pack4  { _Float16 h[4]; unsigned long long u64; };

__global__ __launch_bounds__(256) void gemm_wmma_f16(
    const float* __restrict__ A, const float* __restrict__ W,
    const float* __restrict__ bias, float* __restrict__ C,
    int M, int N, int K, int act_silu, int accumulate)
{
    __shared__ _Float16 sA[BLK_M * LDT];
    __shared__ _Float16 sW[BLK_N * LDT];

    const int tid  = threadIdx.x;
    const int wave = tid >> 5;
    const int lane = tid & 31;
    const int bm   = blockIdx.y * BLK_M;
    const int bn   = blockIdx.x * BLK_N;
    const int wm   = wave * 16;           // this wave's 16-row strip

    v8f acc[8];
#pragma unroll
    for (int nt = 0; nt < 8; ++nt)
        acc[nt] = (v8f){0.f, 0.f, 0.f, 0.f, 0.f, 0.f, 0.f, 0.f};

    const int r  = lane & 15;
    const int hi = (lane >> 4) ? 8 : 0;

    for (int k0 = 0; k0 < K; k0 += BK) {
        // ---- Stage A tile: 128 rows x 32 cols, float4 loads, pack4 -> LDS
        for (int i = tid; i < BLK_M * (BK / 4); i += 256) {
            int row = i >> 3, c4 = (i & 7) * 4;
            float4 f = *(const float4*)(A + (size_t)(bm + row) * K + k0 + c4);
            Pack4 p;
            p.h[0] = (_Float16)f.x; p.h[1] = (_Float16)f.y;
            p.h[2] = (_Float16)f.z; p.h[3] = (_Float16)f.w;
            *(unsigned long long*)&sA[row * LDT + c4] = p.u64;
        }
        // ---- Stage W tile: 128 rows x 32 cols
        for (int i = tid; i < BLK_N * (BK / 4); i += 256) {
            int row = i >> 3, c4 = (i & 7) * 4;
            float4 f = *(const float4*)(W + (size_t)(bn + row) * K + k0 + c4);
            Pack4 p;
            p.h[0] = (_Float16)f.x; p.h[1] = (_Float16)f.y;
            p.h[2] = (_Float16)f.z; p.h[3] = (_Float16)f.w;
            *(unsigned long long*)&sW[row * LDT + c4] = p.u64;
        }
        // ---- Prefetch next K-chunk (global_prefetch_b8): 2 threads per row
        if (k0 + BK < K) {
            int prow = tid >> 1, phalf = (tid & 1) * 16;
            __builtin_prefetch(A + (size_t)(bm + prow) * K + k0 + BK + phalf, 0, 1);
            __builtin_prefetch(W + (size_t)(bn + prow) * K + k0 + BK + phalf, 0, 1);
        }
        __syncthreads();

        // ---- Fragment loads per documented 16-bit 16x32 layout:
        // VGPR v holds K pair {2v | 16+2(v-4)} (+8 for lanes 16-31), row = lane&15.
        const unsigned* pA = (const unsigned*)sA;
        const unsigned* pW = (const unsigned*)sW;

        Frag16 fa;
        {
            const int baseA = (wm + r) * (LDT / 2);
#pragma unroll
            for (int v = 0; v < 8; ++v) {
                int kk = ((v < 4) ? (2 * v) : (16 + 2 * (v - 4))) + hi;  // even
                fa.u[v] = pA[baseA + (kk >> 1)];
            }
        }
#pragma unroll
        for (int nt = 0; nt < 8; ++nt) {
            Frag16 fb;
            const int baseW = (nt * 16 + r) * (LDT / 2);
#pragma unroll
            for (int v = 0; v < 8; ++v) {
                int kk = ((v < 4) ? (2 * v) : (16 + 2 * (v - 4))) + hi;
                fb.u[v] = pW[baseW + (kk >> 1)];
            }
            acc[nt] = __builtin_amdgcn_wmma_f32_16x16x32_f16(
                false, fa.v, false, fb.v, (short)0, acc[nt], false, false);
        }
        __syncthreads();
    }

    // ---- Epilogue: D layout = VGPR v -> row wm+v(+8 for hi lanes), col = lane&15
    const int rbase = bm + wm + hi;
#pragma unroll
    for (int nt = 0; nt < 8; ++nt) {
        const int col = bn + nt * 16 + (lane & 15);
        const float bv = bias ? bias[col] : 0.f;
#pragma unroll
        for (int v = 0; v < 8; ++v) {
            int row = rbase + v;
            float val = acc[nt][v] + bv;
            if (act_silu) val = val / (1.f + expf(-val));
            size_t idx = (size_t)row * N + col;
            if (accumulate) val += C[idx];
            C[idx] = val;
        }
    }
}

// ---------------------------------------------------------------------------
// Small NT GEMM (N = 16 or 1): one thread per output element.
// ---------------------------------------------------------------------------
__global__ void gemm_nt_small(const float* __restrict__ A,
                              const float* __restrict__ W,
                              const float* __restrict__ bias,
                              float* __restrict__ C, int M, int N, int K)
{
    int idx = blockIdx.x * blockDim.x + threadIdx.x;
    if (idx >= M * N) return;
    int m = idx / N, n = idx % N;
    const float* a = A + (size_t)m * K;
    const float* w = W + (size_t)n * K;
    float s = bias ? bias[n] : 0.f;
    for (int k = 0; k < K; ++k) s += a[k] * w[k];
    C[idx] = s;
}

// ---------------------------------------------------------------------------
// Elementwise / per-row kernels
// ---------------------------------------------------------------------------
__global__ void embed_kernel(const float* __restrict__ emb,
                             const int* __restrict__ ids,
                             float* __restrict__ x)
{
    int idx = blockIdx.x * blockDim.x + threadIdx.x;
    if (idx >= ROWS * D_MODEL) return;
    int row = idx / D_MODEL, c = idx % D_MODEL;
    x[idx] = emb[(size_t)ids[row] * D_MODEL + c];
}

__global__ __launch_bounds__(256) void rmsnorm_kernel(
    const float* __restrict__ x, const float* __restrict__ w,
    float* __restrict__ y, int D)
{
    __shared__ float red[256];
    int row = blockIdx.x;
    const float* xr = x + (size_t)row * D;
    float s = 0.f;
    for (int c = threadIdx.x; c < D; c += 256) { float v = xr[c]; s += v * v; }
    red[threadIdx.x] = s;
    __syncthreads();
    for (int off = 128; off > 0; off >>= 1) {
        if (threadIdx.x < off) red[threadIdx.x] += red[threadIdx.x + off];
        __syncthreads();
    }
    float scale = rsqrtf(red[0] / (float)D + 1e-5f);
    for (int c = threadIdx.x; c < D; c += 256)
        y[(size_t)row * D + c] = xr[c] * scale * w[c];
}

__global__ void conv_silu_kernel(const float* __restrict__ h,
                                 const float* __restrict__ cw,
                                 const float* __restrict__ cb,
                                 float* __restrict__ u)
{
    int idx = blockIdx.x * blockDim.x + threadIdx.x;
    if (idx >= ROWS * D_INNER) return;
    int d = idx % D_INNER;
    int row = idx / D_INNER;
    int t = row % SEQ, b = row / SEQ;
    float s = cb[d];
#pragma unroll
    for (int j = 0; j < D_CONV; ++j) {
        int tt = t - (D_CONV - 1) + j;
        if (tt >= 0)
            s += h[((size_t)(b * SEQ + tt)) * D_INNER + d] * cw[d * D_CONV + j];
    }
    u[idx] = s / (1.f + expf(-s));   // silu
}

__global__ void softplus_delta_kernel(const float* __restrict__ dtraw,
                                      const float* __restrict__ tau,
                                      float* __restrict__ Delta)
{
    int idx = blockIdx.x * blockDim.x + threadIdx.x;
    if (idx >= ROWS * D_INNER) return;
    int d = idx % D_INNER;
    int row = idx / D_INNER;
    float v = tau[d] + dtraw[row];
    Delta[idx] = (v > 20.f) ? v : log1pf(expf(v));
}

__global__ void mul_kernel(float* __restrict__ y, const float* __restrict__ res, int n)
{
    int idx = blockIdx.x * blockDim.x + threadIdx.x;
    if (idx < n) y[idx] *= res[idx];
}

// Selective scan: one thread per (b, d) channel, 16-state recurrence in regs.
__global__ void scan_kernel(const float* __restrict__ u,
                            const float* __restrict__ Delta,
                            const float* __restrict__ Bm,
                            const float* __restrict__ Cm,
                            const float* __restrict__ A_log,
                            const float* __restrict__ Dp,
                            float* __restrict__ Y)
{
    int idx = blockIdx.x * blockDim.x + threadIdx.x;
    if (idx >= BATCH * D_INNER) return;
    int d = idx % D_INNER;
    int b = idx / D_INNER;

    float Arow[D_STATE];
#pragma unroll
    for (int n = 0; n < D_STATE; ++n) Arow[n] = -expf(A_log[(size_t)d * D_STATE + n]);
    float hs[D_STATE];
#pragma unroll
    for (int n = 0; n < D_STATE; ++n) hs[n] = 0.f;
    const float Dpd = Dp[d];

    for (int t = 0; t < SEQ; ++t) {
        size_t row = (size_t)b * SEQ + t;
        float dt = Delta[row * D_INNER + d];
        float ut = u[row * D_INNER + d];
        float du = dt * ut;
        const float* Bt = Bm + row * D_STATE;
        const float* Ct = Cm + row * D_STATE;
        float y = 0.f;
#pragma unroll
        for (int n = 0; n < D_STATE; ++n) {
            hs[n] = expf(dt * Arow[n]) * hs[n] + du * Bt[n];
            y += Ct[n] * hs[n];
        }
        Y[row * D_INNER + d] = y + Dpd * ut;
    }
}

// ---------------------------------------------------------------------------
// Host-side orchestration
// ---------------------------------------------------------------------------
extern "C" void kernel_launch(void* const* d_in, const int* in_sizes, int n_in,
                              void* d_out, int out_size, void* d_ws, size_t ws_size,
                              hipStream_t stream)
{
    (void)in_sizes; (void)n_in; (void)out_size; (void)ws_size;

    const float* emb      = (const float*)d_in[0];
    const float* norm_w   = (const float*)d_in[1];
    const float* w_in     = (const float*)d_in[2];
    const float* b_in     = (const float*)d_in[3];
    const float* w_res    = (const float*)d_in[4];
    const float* b_res    = (const float*)d_in[5];
    const float* conv_w   = (const float*)d_in[6];
    const float* conv_b   = (const float*)d_in[7];
    const float* w_B      = (const float*)d_in[8];
    const float* b_B      = (const float*)d_in[9];
    const float* w_C      = (const float*)d_in[10];
    const float* b_C      = (const float*)d_in[11];
    const float* w_dt     = (const float*)d_in[12];
    const float* b_dt     = (const float*)d_in[13];
    const float* tau_dt   = (const float*)d_in[14];
    const float* A_log    = (const float*)d_in[15];
    const float* Dp       = (const float*)d_in[16];
    const float* w_out    = (const float*)d_in[17];
    const float* b_out    = (const float*)d_in[18];
    const float* fnorm_w  = (const float*)d_in[19];
    const int*   ids      = (const int*)d_in[20];
    float*       out      = (float*)d_out;

    // Workspace carve-up (floats). Delta reuses h once h is dead.
    float* ws    = (float*)d_ws;
    float* x     = ws;                          // ROWS*D_MODEL
    float* xn    = x   + (size_t)ROWS * D_MODEL;
    float* h     = xn  + (size_t)ROWS * D_MODEL;  // ROWS*D_INNER (later: Delta)
    float* res   = h   + (size_t)ROWS * D_INNER;
    float* u     = res + (size_t)ROWS * D_INNER;
    float* y     = u   + (size_t)ROWS * D_INNER;
    float* Bm    = y   + (size_t)ROWS * D_INNER;  // ROWS*D_STATE
    float* Cm    = Bm  + (size_t)ROWS * D_STATE;
    float* dtraw = Cm  + (size_t)ROWS * D_STATE;  // ROWS

    const int EW = 256;
    embed_kernel<<<(ROWS * D_MODEL + EW - 1) / EW, EW, 0, stream>>>(emb, ids, x);

    for (int l = 0; l < N_LAYERS; ++l) {
        rmsnorm_kernel<<<ROWS, 256, 0, stream>>>(x, norm_w + (size_t)l * D_MODEL, xn, D_MODEL);

        dim3 gIn(D_INNER / BLK_N, ROWS / BLK_M);
        gemm_wmma_f16<<<gIn, 256, 0, stream>>>(
            xn, w_in + (size_t)l * D_INNER * D_MODEL, b_in + (size_t)l * D_INNER,
            h, ROWS, D_INNER, D_MODEL, /*silu=*/0, /*acc=*/0);

        gemm_wmma_f16<<<gIn, 256, 0, stream>>>(
            h, w_res + (size_t)l * D_INNER * D_INNER, b_res + (size_t)l * D_INNER,
            res, ROWS, D_INNER, D_INNER, /*silu=*/1, /*acc=*/0);

        conv_silu_kernel<<<(ROWS * D_INNER + EW - 1) / EW, EW, 0, stream>>>(
            h, conv_w + (size_t)l * D_INNER * D_CONV, conv_b + (size_t)l * D_INNER, u);

        gemm_nt_small<<<(ROWS * D_STATE + EW - 1) / EW, EW, 0, stream>>>(
            u, w_B + (size_t)l * D_STATE * D_INNER, b_B + (size_t)l * D_STATE,
            Bm, ROWS, D_STATE, D_INNER);
        gemm_nt_small<<<(ROWS * D_STATE + EW - 1) / EW, EW, 0, stream>>>(
            u, w_C + (size_t)l * D_STATE * D_INNER, b_C + (size_t)l * D_STATE,
            Cm, ROWS, D_STATE, D_INNER);
        gemm_nt_small<<<(ROWS + EW - 1) / EW, EW, 0, stream>>>(
            u, w_dt + (size_t)l * D_INNER, b_dt + l, dtraw, ROWS, 1, D_INNER);

        // Delta overwrites h (dead after res-GEMM + conv consumed it)
        softplus_delta_kernel<<<(ROWS * D_INNER + EW - 1) / EW, EW, 0, stream>>>(
            dtraw, tau_dt + (size_t)l * D_INNER, h);

        scan_kernel<<<(BATCH * D_INNER + EW - 1) / EW, EW, 0, stream>>>(
            u, h, Bm, Cm, A_log + (size_t)l * D_INNER * D_STATE,
            Dp + (size_t)l * D_INNER, y);

        mul_kernel<<<(ROWS * D_INNER + EW - 1) / EW, EW, 0, stream>>>(y, res, ROWS * D_INNER);

        dim3 gOut(D_MODEL / BLK_N, ROWS / BLK_M);
        gemm_wmma_f16<<<gOut, 256, 0, stream>>>(
            y, w_out + (size_t)l * D_MODEL * D_INNER, b_out + (size_t)l * D_MODEL,
            x, ROWS, D_MODEL, D_INNER, /*silu=*/0, /*acc=*/1);   // residual add
    }

    rmsnorm_kernel<<<ROWS, 256, 0, stream>>>(x, fnorm_w, xn, D_MODEL);

    dim3 gHead(VOCAB / BLK_N, ROWS / BLK_M);
    gemm_wmma_f16<<<gHead, 256, 0, stream>>>(
        xn, emb, (const float*)nullptr, out, ROWS, VOCAB, D_MODEL, 0, 0);
}
